// FFT_75677323756101
// MI455X (gfx1250) — matrix operand order repaired and checked by reading, not compiled, not run on hardware
//
#include <hip/hip_runtime.h>

typedef __attribute__((ext_vector_type(16))) __bf16       v16bf;
typedef __attribute__((ext_vector_type(8)))  float        v8f;
typedef __attribute__((ext_vector_type(4)))  unsigned int u32x4;

#define M_TILE  128
#define N_TILE  64
#define K_TILE  32
#define LDA     40      // bf16 elems per LDS row: 32 + 8 pad (80B = 20 banks, conflict-free b128 reads)
#define LDB     40
#define K_STEPS 32      // 1024 / 32
#define EPAD    68      // f32 epilogue LDS stride (conflict-free)
#define NCOLS   1024
#define OUT_HALF ((size_t)32768 * 1024)

union Frag {
    v16bf v;
    u32x4 u[2];
};

struct alignas(8) bh4 { __bf16 x, y, z, w; };

// One pipeline stage: all six matrices grouped so every ds_load/ds_store within the
// active stage is reachable via the 16-bit immediate offset (stage size 46080 B < 64K).
struct Stage {
    __bf16 Ar[M_TILE * LDA];   // bf16(xr)
    __bf16 Ai[M_TILE * LDA];   // bf16(xi)
    __bf16 As[M_TILE * LDA];   // bf16(xr + xi)
    __bf16 Br[N_TILE * LDB];   // bf16(wr)
    __bf16 Bi[N_TILE * LDB];   // bf16(wi)
    __bf16 Bs[N_TILE * LDB];   // bf16(wr + wi)
};

__device__ __forceinline__ void cvt_store4(__bf16* p, float a, float b, float c, float d) {
    bh4 t; t.x = (__bf16)a; t.y = (__bf16)b; t.z = (__bf16)c; t.w = (__bf16)d;
    *reinterpret_cast<bh4*>(p) = t;      // 8-byte DS store of 4 packed bf16
}

// A fragment 16x32 bf16: lanes 0-15 -> row M=lane, K {0..7,16..23}; lanes 16-31 -> K {8..15,24..31}
__device__ __forceinline__ v16bf frag_a(const __bf16* base, int row0, int lane) {
    const int half = lane >> 4;
    const __bf16* p = base + (row0 + (lane & 15)) * LDA + half * 8;
    Frag f;
    f.u[0] = *reinterpret_cast<const u32x4*>(p);        // K = kh .. kh+7
    f.u[1] = *reinterpret_cast<const u32x4*>(p + 16);   // K = kh+16 .. kh+23
    return f.v;
}

// B fragment 32x16 bf16: lanes 0-15 -> col N=lane, K {0..7,16..23}; lanes 16-31 -> K {8..15,24..31}
__device__ __forceinline__ v16bf frag_b(const __bf16* base, int col0, int lane) {
    const int half = lane >> 4;
    const __bf16* p = base + (col0 + (lane & 15)) * LDB + half * 8;
    Frag f;
    f.u[0] = *reinterpret_cast<const u32x4*>(p);        // K = kh .. kh+7
    f.u[1] = *reinterpret_cast<const u32x4*>(p + 16);   // K = kh+16 .. kh+23
    return f.v;
}

__global__ __launch_bounds__(256)
void dft_wmma_kernel(const float* __restrict__ xr, const float* __restrict__ xi,
                     const float* __restrict__ wr, const float* __restrict__ wi,
                     const float* __restrict__ br, const float* __restrict__ bi,
                     float* __restrict__ outR, float* __restrict__ outI)
{
    __shared__ union {
        Stage st[2];                      // 92160 B double-buffered stage
        float E[M_TILE * EPAD];           // 34816 B epilogue staging
    } sm;

    const int tid  = threadIdx.x;
    const int lane = tid & 31;
    const int wv   = tid >> 5;            // wave 0..7
    const int mw   = (wv >> 1) * 32;      // wave's M offset in tile
    const int nw   = (wv & 1) * 32;       // wave's N offset in tile

    const int gm0 = (int)(blockIdx.x >> 4) * M_TILE;   // 256 row-blocks
    const int gn0 = (int)(blockIdx.x & 15) * N_TILE;   // 16 col-blocks (consecutive blocks share A stripe -> L2)

    const int arow = tid >> 3;            // 0..31 (row group for staging loads)
    const int ac4  = (tid & 7) * 4;       // float4 column within 32-wide K slab

    // 32-bit loop-carried element offsets; row groups differ by compile-time constants
    // that fold into the 24-bit global-load immediate offset.
    unsigned offA = (unsigned)(gm0 + arow) * NCOLS + (unsigned)ac4;
    unsigned offB = (unsigned)(gn0 + arow) * NCOLS + (unsigned)ac4;

    float4 ra[4], ria[4], rb[2], rib[2];  // register prefetch staging

    auto load_stage = [&]() {
        #pragma unroll
        for (int i = 0; i < 4; ++i) {
            ra[i]  = *reinterpret_cast<const float4*>(xr + offA + (size_t)(i * 32 * NCOLS));
            ria[i] = *reinterpret_cast<const float4*>(xi + offA + (size_t)(i * 32 * NCOLS));
        }
        #pragma unroll
        for (int i = 0; i < 2; ++i) {
            rb[i]  = *reinterpret_cast<const float4*>(wr + offB + (size_t)(i * 32 * NCOLS));
            rib[i] = *reinterpret_cast<const float4*>(wi + offB + (size_t)(i * 32 * NCOLS));
        }
        offA += K_TILE;   // advance to next 32-wide K slab
        offB += K_TILE;
    };

    auto store_stage = [&](Stage* g) {
        #pragma unroll
        for (int i = 0; i < 4; ++i) {
            const int r = arow + i * 32;
            cvt_store4(&g->Ar[r * LDA + ac4], ra[i].x,  ra[i].y,  ra[i].z,  ra[i].w);
            cvt_store4(&g->Ai[r * LDA + ac4], ria[i].x, ria[i].y, ria[i].z, ria[i].w);
            cvt_store4(&g->As[r * LDA + ac4], ra[i].x + ria[i].x, ra[i].y + ria[i].y,
                                              ra[i].z + ria[i].z, ra[i].w + ria[i].w);
        }
        #pragma unroll
        for (int i = 0; i < 2; ++i) {
            const int n = arow + i * 32;
            cvt_store4(&g->Br[n * LDB + ac4], rb[i].x,  rb[i].y,  rb[i].z,  rb[i].w);
            cvt_store4(&g->Bi[n * LDB + ac4], rib[i].x, rib[i].y, rib[i].z, rib[i].w);
            cvt_store4(&g->Bs[n * LDB + ac4], rb[i].x + rib[i].x, rb[i].y + rib[i].y,
                                              rb[i].z + rib[i].z, rb[i].w + rib[i].w);
        }
    };

    // Gauss/Karatsuba 3-multiply complex product accumulators:
    v8f acc1[2][2] = {};   // t1 = xr@wr
    v8f acc2[2][2] = {};   // t2 = xi@wi
    v8f acc3[2][2] = {};   // t3 = (xr+xi)@(wr+wi)
    // real = t1 - t2 ; imag = t3 - t1 - t2

    load_stage();
    store_stage(&sm.st[0]);
    __syncthreads();

    for (int k = 0; k < K_STEPS; ++k) {
        const Stage* g = &sm.st[k & 1];
        if (k + 1 < K_STEPS) load_stage();   // global loads overlap WMMA below

        v16bf a_r[2], a_i[2], a_s[2];
        #pragma unroll
        for (int mt = 0; mt < 2; ++mt) {
            a_r[mt] = frag_a(g->Ar, mw + mt * 16, lane);
            a_i[mt] = frag_a(g->Ai, mw + mt * 16, lane);
            a_s[mt] = frag_a(g->As, mw + mt * 16, lane);
        }
        #pragma unroll
        for (int nt = 0; nt < 2; ++nt) {
            v16bf b_r = frag_b(g->Br, nw + nt * 16, lane);
            v16bf b_i = frag_b(g->Bi, nw + nt * 16, lane);
            v16bf b_s = frag_b(g->Bs, nw + nt * 16, lane);
            #pragma unroll
            for (int mt = 0; mt < 2; ++mt) {
                acc1[mt][nt] = __builtin_amdgcn_wmma_f32_16x16x32_bf16(false, a_r[mt], false, b_r, (short)0, acc1[mt][nt], false, false);
                acc2[mt][nt] = __builtin_amdgcn_wmma_f32_16x16x32_bf16(false, a_i[mt], false, b_i, (short)0, acc2[mt][nt], false, false);
                acc3[mt][nt] = __builtin_amdgcn_wmma_f32_16x16x32_bf16(false, a_s[mt], false, b_s, (short)0, acc3[mt][nt], false, false);
            }
        }

        if (k + 1 < K_STEPS) store_stage(&sm.st[(k + 1) & 1]);
        __syncthreads();
    }

    // ---- Epilogue: stage combined results in LDS, then coalesced float4 stores with bias ----
    const int l16 = lane & 15;
    const int l8  = (lane >> 4) * 8;   // C/D layout: VGPR v holds M=v (lanes 0-15) / M=v+8 (lanes 16-31)

    // pass 1: real = t1 - t2 + (b_real - b_imag)
    #pragma unroll
    for (int mt = 0; mt < 2; ++mt)
        #pragma unroll
        for (int nt = 0; nt < 2; ++nt)
            #pragma unroll
            for (int v = 0; v < 8; ++v)
                sm.E[(mw + mt * 16 + v + l8) * EPAD + nw + nt * 16 + l16] =
                    acc1[mt][nt][v] - acc2[mt][nt][v];
    __syncthreads();
    #pragma unroll
    for (int i = 0; i < 8; ++i) {
        const int idx = tid + i * 256;
        const int row = idx >> 4;
        const int c4  = (idx & 15) * 4;
        const float4 v  = *reinterpret_cast<const float4*>(&sm.E[row * EPAD + c4]);
        const float4 b1 = *reinterpret_cast<const float4*>(br + gn0 + c4);
        const float4 b2 = *reinterpret_cast<const float4*>(bi + gn0 + c4);
        float4 o;
        o.x = v.x + b1.x - b2.x;  o.y = v.y + b1.y - b2.y;
        o.z = v.z + b1.z - b2.z;  o.w = v.w + b1.w - b2.w;
        *reinterpret_cast<float4*>(outR + (size_t)(gm0 + row) * NCOLS + gn0 + c4) = o;
    }
    __syncthreads();

    // pass 2: imag = t3 - t1 - t2 + (b_real + b_imag)
    #pragma unroll
    for (int mt = 0; mt < 2; ++mt)
        #pragma unroll
        for (int nt = 0; nt < 2; ++nt)
            #pragma unroll
            for (int v = 0; v < 8; ++v)
                sm.E[(mw + mt * 16 + v + l8) * EPAD + nw + nt * 16 + l16] =
                    acc3[mt][nt][v] - acc1[mt][nt][v] - acc2[mt][nt][v];
    __syncthreads();
    #pragma unroll
    for (int i = 0; i < 8; ++i) {
        const int idx = tid + i * 256;
        const int row = idx >> 4;
        const int c4  = (idx & 15) * 4;
        const float4 v  = *reinterpret_cast<const float4*>(&sm.E[row * EPAD + c4]);
        const float4 b1 = *reinterpret_cast<const float4*>(br + gn0 + c4);
        const float4 b2 = *reinterpret_cast<const float4*>(bi + gn0 + c4);
        float4 o;
        o.x = v.x + b1.x + b2.x;  o.y = v.y + b1.y + b2.y;
        o.z = v.z + b1.z + b2.z;  o.w = v.w + b1.w + b2.w;
        *reinterpret_cast<float4*>(outI + (size_t)(gm0 + row) * NCOLS + gn0 + c4) = o;
    }
}

extern "C" void kernel_launch(void* const* d_in, const int* in_sizes, int n_in,
                              void* d_out, int out_size, void* d_ws, size_t ws_size,
                              hipStream_t stream) {
    const float* xr = (const float*)d_in[0];
    const float* xi = (const float*)d_in[1];
    const float* wr = (const float*)d_in[2];
    const float* wi = (const float*)d_in[3];
    const float* br = (const float*)d_in[4];
    const float* bi = (const float*)d_in[5];
    float* outR = (float*)d_out;
    float* outI = outR + OUT_HALF;

    dim3 grid(4096), block(256);   // (32768/128) * (1024/64) workgroups
    hipLaunchKernelGGL(dft_wmma_kernel, grid, block, 0, stream,
                       xr, xi, wr, wi, br, bi, outR, outI);
}
